// MultiHeadCrossAttention_27169963114617
// MI455X (gfx1250) — compile-verified, matrix operand order checked by reading, not run on hardware
//
#include <hip/hip_runtime.h>
#include <hip/hip_bf16.h>
#include <stdint.h>

typedef __attribute__((ext_vector_type(16))) __bf16 v16bf;
typedef __attribute__((ext_vector_type(8)))  float  v8f;
typedef __attribute__((ext_vector_type(4)))  int    v4i;

#define EDIM 512
#define NH   8
#define HD   64
#define SEQ  4096
#define NBAT 2
#define SM_SCALE 0.125f

union ABu { v16bf v; unsigned u[8]; };

// K-pair base for 16-bit A/B operand: dword g, lane-half hi (0: lanes0-15, 1: lanes16-31)
__device__ __forceinline__ int kp(int g, int hi) {
  return ((g & 3) << 1) + ((g >> 2) << 4) + (hi << 3);
}

__device__ __forceinline__ unsigned short f2bf(float f) {
  unsigned u = __builtin_bit_cast(unsigned, f);
  u += 0x7FFFu + ((u >> 16) & 1u);          // round-to-nearest-even
  return (unsigned short)(u >> 16);
}

#if defined(__has_builtin)
#if __has_builtin(__builtin_amdgcn_global_load_async_to_lds_b128) && \
    __has_builtin(__builtin_amdgcn_s_wait_asynccnt)
#define USE_ASYNC 1
#endif
#endif

typedef __attribute__((address_space(1))) v4i GV4;   // global int4
typedef __attribute__((address_space(3))) v4i LV4;   // LDS int4

// ---------------------------------------------------------------- kernel 0
__global__ __launch_bounds__(256) void wconv_kernel(const float* __restrict__ w,
                                                    unsigned short* __restrict__ wbf) {
  int i = (blockIdx.x * 256 + threadIdx.x) * 4;
#pragma unroll
  for (int j = 0; j < 4; ++j) wbf[i + j] = f2bf(w[i + j]);
}

// ---------------------------------------------------------------- kernel 1
// MODE 0: input x  -> K [B,H,N,64] bf16, Vt [B,H,64,N] bf16
// MODE 1: input y  -> Q [B,H,M,64] bf16, yn fp32 (residual)
template <int MODE>
__global__ __launch_bounds__(256) void ln_qkv_kernel(
    const float* __restrict__ inp, const float* __restrict__ lnw,
    const float* __restrict__ lnb, const unsigned* __restrict__ wbf32,
    unsigned short* __restrict__ Qb, unsigned short* __restrict__ Kb,
    unsigned short* __restrict__ Vtb, float* __restrict__ yn) {
  __shared__ __align__(16) unsigned short Xn[16][520];  // 16 rows LN'd input, bf16
  const int tid = threadIdx.x, w = tid >> 5, lane = tid & 31;
  const int hi = lane >> 4, ln16 = lane & 15;
  const int row0 = blockIdx.x * 16;

  // ---- Phase A: LayerNorm, one wave per row (2 rows/wave), shuffle reduce
  for (int rr = 0; rr < 2; ++rr) {
    const int r = w * 2 + rr;
    const int gr = row0 + r;
    float v[16], s = 0.f, q = 0.f;
#pragma unroll
    for (int i = 0; i < 16; ++i) {
      float t = inp[(size_t)gr * EDIM + lane + 32 * i];
      v[i] = t; s += t; q += t * t;
    }
#pragma unroll
    for (int m = 16; m >= 1; m >>= 1) {
      s += __shfl_xor(s, m, 32);
      q += __shfl_xor(q, m, 32);
    }
    const float mean = s * (1.f / EDIM);
    const float rstd = rsqrtf(q * (1.f / EDIM) - mean * mean + 1e-5f);
#pragma unroll
    for (int i = 0; i < 16; ++i) {
      const int e = lane + 32 * i;
      const float xv = (v[i] - mean) * rstd * lnw[e] + lnb[e];
      Xn[r][e] = f2bf(xv);
      if (MODE == 1) yn[(size_t)gr * EDIM + e] = xv;
    }
  }
  __syncthreads();

  // ---- Phase B: [16 x 512] @ W^T tile via WMMA; wave owns NT col-tiles
  constexpr int NT = (MODE == 0) ? 8 : 4;
  v8f acc[NT];
#pragma unroll
  for (int t = 0; t < NT; ++t) acc[t] = (v8f)0.f;

  const unsigned* Xn32 = (const unsigned*)&Xn[0][0];
  for (int ek = 0; ek < 16; ++ek) {            // K loop: 512 = 16 x 32
    ABu a;
#pragma unroll
    for (int g = 0; g < 8; ++g) {
      const int k = ek * 32 + kp(g, hi);
      a.u[g] = Xn32[ln16 * 260 + (k >> 1)];
    }
#pragma unroll
    for (int t = 0; t < NT; ++t) {
      const int jrow = ((MODE == 0) ? 512 : 0) + (w * NT + t) * 16 + ln16;
      ABu bb;
#pragma unroll
      for (int g = 0; g < 8; ++g) {
        const int k = ek * 32 + kp(g, hi);
        bb.u[g] = wbf32[jrow * 256 + (k >> 1)];
      }
      acc[t] = __builtin_amdgcn_wmma_f32_16x16x32_bf16(
          false, a.v, false, bb.v, (short)0, acc[t], false, false);
    }
  }

  // ---- store heads-aware
#pragma unroll
  for (int t = 0; t < NT; ++t) {
#pragma unroll
    for (int g = 0; g < 8; ++g) {
      const int m = g + 8 * hi;
      const int gr = row0 + m;
      const int b = gr >> 12, n = gr & (SEQ - 1);
      const unsigned short val = f2bf(acc[t][g]);
      if (MODE == 1) {
        const int j = (w * NT + t) * 16 + ln16;       // 0..511 -> Q
        const int h = j >> 6, d = j & 63;
        Qb[(((size_t)(b * NH + h) * SEQ + n) << 6) + d] = val;
      } else if (w < 4) {
        const int j = (w * NT + t) * 16 + ln16;       // 0..511 -> K
        const int h = j >> 6, d = j & 63;
        Kb[(((size_t)(b * NH + h) * SEQ + n) << 6) + d] = val;
      } else {
        const int j = (w * NT + t) * 16 + ln16 - 512; // 0..511 -> V (transposed)
        const int h = j >> 6, d = j & 63;
        Vtb[(((size_t)(b * NH + h) * HD + d) << 12) + n] = val;
      }
    }
  }
}

// ---------------------------------------------------------------- kernel 2
// Flash attention: block = (b,h, 64 Q rows); loop over 64-row K/V tiles.
// K/V tiles double-buffered in LDS; tile nb+1 is async-prefetched while
// tile nb is consumed (s_wait_asynccnt 4 = oldest tile's 4 loads complete).
__global__ __launch_bounds__(256) void attn_kernel(
    const unsigned short* __restrict__ Qb, const unsigned short* __restrict__ Kb,
    const unsigned short* __restrict__ Vtb, const float* __restrict__ yn,
    float* __restrict__ out) {
  __shared__ __align__(16) unsigned short Kt[2][64][72];  // [buf][n][d]
  __shared__ __align__(16) unsigned short Vt[2][64][72];  // [buf][d][n]
  __shared__ __align__(16) unsigned short Pt[64][72];     // [m][n] bf16 probs
  __shared__ float Srow[64][66];                          // fp32 scores staging
  __shared__ float mrow[64], lrow[64], arow[64];

  const int tid = threadIdx.x, w = tid >> 5, lane = tid & 31;
  const int hi = lane >> 4, ln16 = lane & 15;
  const int b = blockIdx.z, h = blockIdx.y, m0 = blockIdx.x * 64;
  const int bh = b * NH + h;
  const int mtile = w & 3, pair = w >> 2;
  constexpr int NB = SEQ / 64;

  if (tid < 64) { mrow[tid] = -1e30f; lrow[tid] = 0.f; }

  // ---- staging: issue 4 async-load (or sync copy) instrs per wave for tile nb
  const int sr = tid >> 3, sc = (tid & 7) * 8;   // 32 rows x 8 chunks per pass
  auto stage = [&](int nb, int buf) {
#pragma unroll
    for (int it = 0; it < 2; ++it) {
      const int r = sr + it * 32, c = sc;
      const uint4* gk = (const uint4*)(Kb  + (((size_t)bh * SEQ + nb * 64 + r) << 6) + c);
      const uint4* gv = (const uint4*)(Vtb + (((size_t)bh * HD + r) << 12) + nb * 64 + c);
#ifdef USE_ASYNC
      __builtin_amdgcn_global_load_async_to_lds_b128((GV4*)gk, (LV4*)&Kt[buf][r][c], 0, 0);
      __builtin_amdgcn_global_load_async_to_lds_b128((GV4*)gv, (LV4*)&Vt[buf][r][c], 0, 0);
#else
      *(uint4*)&Kt[buf][r][c] = *gk;
      *(uint4*)&Vt[buf][r][c] = *gv;
#endif
    }
  };

  // Q strip (16 rows x 64) as two A operands, kept in registers
  ABu qa[2];
#pragma unroll
  for (int c = 0; c < 2; ++c) {
#pragma unroll
    for (int g = 0; g < 8; ++g) {
      const int row = m0 + mtile * 16 + ln16;
      const int k = c * 32 + kp(g, hi);
      qa[c].u[g] = ((const unsigned*)Qb)[(((size_t)bh * SEQ + row) << 5) + (k >> 1)];
    }
  }

  v8f o[2];
  o[0] = (v8f)0.f; o[1] = (v8f)0.f;
  const unsigned* Pt32 = (const unsigned*)&Pt[0][0];

  stage(0, 0);

  for (int nb = 0; nb < NB; ++nb) {
    const int cur = nb & 1;
    const bool pre = (nb + 1 < NB);
    if (pre) stage(nb + 1, cur ^ 1);          // prefetch next tile into other buffer
#ifdef USE_ASYNC
    if (pre) __builtin_amdgcn_s_wait_asynccnt(4);   // current tile's 4 loads done
    else     __builtin_amdgcn_s_wait_asynccnt(0);
#endif
    __syncthreads();

    const unsigned* Kt32 = (const unsigned*)&Kt[cur][0][0];
    const unsigned* Vt32 = (const unsigned*)&Vt[cur][0][0];

    // ---- S = Q K^T * scale  (each wave: 16x32 of the 64x64 tile)
#pragma unroll
    for (int t = 0; t < 2; ++t) {
      const int nt = pair * 2 + t;
      v8f s = (v8f)0.f;
#pragma unroll
      for (int c = 0; c < 2; ++c) {
        ABu bb;
#pragma unroll
        for (int g = 0; g < 8; ++g) {
          const int k = c * 32 + kp(g, hi);
          bb.u[g] = Kt32[(nt * 16 + ln16) * 36 + (k >> 1)];
        }
        s = __builtin_amdgcn_wmma_f32_16x16x32_bf16(
            false, qa[c].v, false, bb.v, (short)0, s, false, false);
      }
#pragma unroll
      for (int g = 0; g < 8; ++g)
        Srow[mtile * 16 + g + 8 * hi][nt * 16 + ln16] = s[g] * SM_SCALE;
    }
    __syncthreads();

    // ---- online softmax: 2 threads per row, partner via shfl_xor(1)
    if (tid < 128) {
      const int r = tid >> 1, half = tid & 1, c0 = half * 32;
      float mx = -1e30f;
#pragma unroll 8
      for (int c = 0; c < 32; ++c) mx = fmaxf(mx, Srow[r][c0 + c]);
      mx = fmaxf(mx, __shfl_xor(mx, 1, 32));
      const float mold = mrow[r];
      const float mnew = fmaxf(mold, mx);
      float sum = 0.f;
#pragma unroll 8
      for (int c = 0; c < 32; ++c) {
        const float p = __expf(Srow[r][c0 + c] - mnew);
        Pt[r][c0 + c] = f2bf(p);
        sum += p;
      }
      sum += __shfl_xor(sum, 1, 32);
      if (half == 0) {
        const float alpha = __expf(mold - mnew);
        arow[r] = alpha;
        mrow[r] = mnew;
        lrow[r] = alpha * lrow[r] + sum;
      }
    }
    __syncthreads();

    // ---- rescale accumulator, O += P * V
#pragma unroll
    for (int g = 0; g < 8; ++g) {
      const float a = arow[mtile * 16 + g + 8 * hi];
      o[0][g] *= a; o[1][g] *= a;
    }
#pragma unroll
    for (int c = 0; c < 2; ++c) {
      ABu pa;
#pragma unroll
      for (int g = 0; g < 8; ++g) {
        const int k = c * 32 + kp(g, hi);
        pa.u[g] = Pt32[(mtile * 16 + ln16) * 36 + (k >> 1)];
      }
#pragma unroll
      for (int t = 0; t < 2; ++t) {
        const int dt = pair * 2 + t;
        ABu vb;
#pragma unroll
        for (int g = 0; g < 8; ++g) {
          const int k = c * 32 + kp(g, hi);
          vb.u[g] = Vt32[(dt * 16 + ln16) * 36 + (k >> 1)];
        }
        o[t] = __builtin_amdgcn_wmma_f32_16x16x32_bf16(
            false, pa.v, false, vb.v, (short)0, o[t], false, false);
      }
    }
    __syncthreads();
  }

  // ---- epilogue: normalize, add residual yn, store fp32
#pragma unroll
  for (int g = 0; g < 8; ++g) {
    const int m = mtile * 16 + g + 8 * hi;
    const float inv = 1.f / lrow[m];
    const int gm = m0 + m;
#pragma unroll
    for (int t = 0; t < 2; ++t) {
      const int d = (pair * 2 + t) * 16 + ln16;
      const size_t oi = ((size_t)(b * SEQ + gm)) * EDIM + h * HD + d;
      out[oi] = o[t][g] * inv + yn[oi];
    }
  }
}

// ---------------------------------------------------------------- launch
extern "C" void kernel_launch(void* const* d_in, const int* in_sizes, int n_in,
                              void* d_out, int out_size, void* d_ws, size_t ws_size,
                              hipStream_t stream) {
  const float* x    = (const float*)d_in[0];
  const float* y    = (const float*)d_in[1];
  const float* ln_w = (const float*)d_in[2];
  const float* ln_b = (const float*)d_in[3];
  const float* qkvw = (const float*)d_in[4];
  float* out = (float*)d_out;

  char* ws = (char*)d_ws;
  const size_t wbf_bytes = (size_t)3 * EDIM * EDIM * 2;            // 1.5 MB
  const size_t qkv_bytes = (size_t)NBAT * NH * SEQ * HD * 2;       // 8 MB each
  unsigned short* Wbf = (unsigned short*)ws;
  unsigned short* Qb  = (unsigned short*)(ws + wbf_bytes);
  unsigned short* Kb  = (unsigned short*)(ws + wbf_bytes + qkv_bytes);
  unsigned short* Vtb = (unsigned short*)(ws + wbf_bytes + 2 * qkv_bytes);
  float* yn           = (float*)(ws + wbf_bytes + 3 * qkv_bytes);

  wconv_kernel<<<(3 * EDIM * EDIM) / 1024, 256, 0, stream>>>(qkvw, Wbf);

  const int rows = NBAT * SEQ;
  ln_qkv_kernel<0><<<rows / 16, 256, 0, stream>>>(x, ln_w, ln_b, (const unsigned*)Wbf,
                                                  Qb, Kb, Vtb, yn);
  ln_qkv_kernel<1><<<rows / 16, 256, 0, stream>>>(y, ln_w, ln_b, (const unsigned*)Wbf,
                                                  Qb, Kb, Vtb, yn);

  attn_kernel<<<dim3(SEQ / 64, NH, NBAT), 256, 0, stream>>>(Qb, Kb, Vtb, yn, out);
}
